// QKNormAttention_29317446762603
// MI455X (gfx1250) — matrix operand order repaired and checked by reading, not compiled, not run on hardware
//
#include <hip/hip_runtime.h>

typedef __attribute__((ext_vector_type(16))) __bf16 v16bf;
typedef __attribute__((ext_vector_type(8)))  float  v8f;
typedef unsigned short u16;
typedef unsigned int   u32;

#define DEV __device__ __forceinline__

static constexpr int BATCH = 4;
static constexpr int SEQ   = 2048;
static constexpr int DIM   = 1024;
static constexpr int HEADS = 16;
static constexpr int HD    = 64;
static constexpr int NQKV  = 3 * DIM;            // 3072
static constexpr int MTOT  = BATCH * SEQ;        // 8192
static constexpr int ROWS_PER_BUF = BATCH * HEADS * SEQ;  // 131072

DEV u16 f2bf(float f) {
  u32 x = __float_as_uint(f);
  x += 0x7fffu + ((x >> 16) & 1u);               // round-to-nearest-even
  return (u16)(x >> 16);
}
DEV float bf2f(u16 h) { return __uint_as_float(((u32)h) << 16); }

union AF { v16bf v; u32 u[8]; };

// A-matrix 16x32 bf16 fragment (ISA 7.12.2): VGPR j, lane-half h:
// K = (j<4?0:16) + h*8 + 2*(j&3) (+1 in high dword half). Row = lane&15.
DEV v16bf load_afrag(const u16* lds, int rowIdx, int stride, int kBase, int lane) {
  const int half = (lane >> 4) & 1;
  const u16* p = lds + rowIdx * stride;
  AF f;
#pragma unroll
  for (int j = 0; j < 8; ++j) {
    int kk = kBase + ((j < 4) ? 0 : 16) + half * 8 + 2 * (j & 3);
    f.u[j] = *(const u32*)(p + kk);
  }
  return f.v;
}

// B-matrix 32x16 bf16 fragment, LDS stored [n][k] (k contiguous):
// VGPR j holds K = (j<4?0:16) + half*8 + 2*(j&3) (+1), N = lane&15.
DEV v16bf load_bfrag(const u16* lds, int nBase, int stride, int kBase, int lane) {
  const int half = (lane >> 4) & 1;
  const int row  = lane & 15;
  const u16* p = lds + (nBase + row) * stride;
  AF f;
#pragma unroll
  for (int j = 0; j < 8; ++j) {
    int kk = kBase + ((j < 4) ? 0 : 16) + half * 8 + 2 * (j & 3);
    f.u[j] = *(const u32*)(p + kk);
  }
  return f.v;
}

DEV v8f wmma_bf16(v16bf a, v16bf b, v8f c) {
  return __builtin_amdgcn_wmma_f32_16x16x32_bf16(false, a, false, b, (short)0, c,
                                                 false, false);
}

// ---------------------------------------------------------------------------
// Kernel 1: qkv = x @ qkv_w^T + b, scattered into q/k/v [B,H,S,Hd] bf16
// Block 256 = 8 waves (2x4); block tile 64(M) x 64(N); K-step 64.
// Each wave: two 16x16 M-tiles -> 4 WMMAs per barrier interval.
// ---------------------------------------------------------------------------
__global__ __launch_bounds__(256) void qkv_gemm_kernel(
    const float* __restrict__ x, const float* __restrict__ w,
    const float* __restrict__ bias,
    u16* __restrict__ qb, u16* __restrict__ kb, u16* __restrict__ vb) {
  __shared__ u16 la[64 * 72];
  __shared__ u16 lb[64 * 72];
  const int tid = threadIdx.x, lane = tid & 31, wv = tid >> 5;
  const int wr = wv >> 2, wc = wv & 3;
  const int m0 = blockIdx.x * 64, n0 = blockIdx.y * 64;

  v8f acc0 = {0.f,0.f,0.f,0.f,0.f,0.f,0.f,0.f};
  v8f acc1 = {0.f,0.f,0.f,0.f,0.f,0.f,0.f,0.f};

  for (int k0 = 0; k0 < DIM; k0 += 64) {
    if (k0 + 64 < DIM) {  // prefetch next K tiles (one cacheline per thread)
      __builtin_prefetch(x + (size_t)(m0 + (tid >> 2)) * DIM + k0 + 64 + (tid & 3) * 16, 0, 0);
      __builtin_prefetch(w + (size_t)(n0 + (tid >> 2)) * DIM + k0 + 64 + (tid & 3) * 16, 0, 0);
    }
#pragma unroll
    for (int r = 0; r < 4; ++r) {  // stage A: 64x64 fp32 -> bf16
      int e = tid * 4 + r * 1024, row = e >> 6, col = e & 63;
      float4 f = *(const float4*)(x + (size_t)(m0 + row) * DIM + k0 + col);
      uint2 pk;
      pk.x = (u32)f2bf(f.x) | ((u32)f2bf(f.y) << 16);
      pk.y = (u32)f2bf(f.z) | ((u32)f2bf(f.w) << 16);
      *(uint2*)(la + row * 72 + col) = pk;
    }
#pragma unroll
    for (int r = 0; r < 4; ++r) {  // stage B: 64(n) x 64(k), [n][k]
      int e = tid * 4 + r * 1024, row = e >> 6, col = e & 63;
      float4 f = *(const float4*)(w + (size_t)(n0 + row) * DIM + k0 + col);
      uint2 pk;
      pk.x = (u32)f2bf(f.x) | ((u32)f2bf(f.y) << 16);
      pk.y = (u32)f2bf(f.z) | ((u32)f2bf(f.w) << 16);
      *(uint2*)(lb + row * 72 + col) = pk;
    }
    __syncthreads();
#pragma unroll
    for (int ks = 0; ks < 64; ks += 32) {
      v16bf b  = load_bfrag(lb, wc * 16, 72, ks, lane);
      v16bf a0 = load_afrag(la, wr * 32 + (lane & 15), 72, ks, lane);
      v16bf a1 = load_afrag(la, wr * 32 + 16 + (lane & 15), 72, ks, lane);
      acc0 = wmma_bf16(a0, b, acc0);
      acc1 = wmma_bf16(a1, b, acc1);
    }
    __syncthreads();
  }

  const int half = lane >> 4;
  const int ncol = n0 + wc * 16 + (lane & 15);
  const float bsv = bias[ncol];
  const int which = ncol >> 10;        // 0=q, 1=k, 2=v
  const int dmod = ncol & 1023;
  const int h = dmod >> 6, hd = dmod & 63;
  u16* dst = (which == 0) ? qb : ((which == 1) ? kb : vb);
#pragma unroll
  for (int e = 0; e < 8; ++e) {
    int m = m0 + wr * 32 + e + 8 * half;          // C/D layout: row = e+8*half
    int bi = m >> 11, s = m & 2047;
    dst[((size_t)(bi * HEADS + h) * SEQ + s) * HD + hd] = f2bf(acc0[e] + bsv);
    m += 16; bi = m >> 11; s = m & 2047;
    dst[((size_t)(bi * HEADS + h) * SEQ + s) * HD + hd] = f2bf(acc1[e] + bsv);
  }
}

// ---------------------------------------------------------------------------
// Kernel 2: in-place LayerNorm over head_dim=64 for q and k (one wave/row)
// ---------------------------------------------------------------------------
__global__ __launch_bounds__(256) void qk_ln_kernel(
    u16* __restrict__ qb, u16* __restrict__ kb,
    const float* __restrict__ qg, const float* __restrict__ qbt,
    const float* __restrict__ kg, const float* __restrict__ kbt) {
  const int lane = threadIdx.x & 31;
  const int rid = blockIdx.x * 8 + (threadIdx.x >> 5);
  u16* row;
  const float *g, *bt;
  if (rid < ROWS_PER_BUF) { row = qb + (size_t)rid * HD; g = qg; bt = qbt; }
  else { row = kb + (size_t)(rid - ROWS_PER_BUF) * HD; g = kg; bt = kbt; }

  u32 u = ((const u32*)row)[lane];
  float x0 = bf2f((u16)(u & 0xffff)), x1 = bf2f((u16)(u >> 16));
  float sm = x0 + x1, sq = x0 * x0 + x1 * x1;
#pragma unroll
  for (int mask = 16; mask >= 1; mask >>= 1) {
    sm += __shfl_xor(sm, mask, 32);
    sq += __shfl_xor(sq, mask, 32);
  }
  const float mean = sm * (1.0f / 64.0f);
  float var = sq * (1.0f / 64.0f) - mean * mean;
  var = fmaxf(var, 0.0f);
  const float rstd = rsqrtf(var + 1e-5f);
  float y0 = (x0 - mean) * rstd * g[2 * lane]     + bt[2 * lane];
  float y1 = (x1 - mean) * rstd * g[2 * lane + 1] + bt[2 * lane + 1];
  ((u32*)row)[lane] = (u32)f2bf(y0) | ((u32)f2bf(y1) << 16);
}

// ---------------------------------------------------------------------------
// Kernel 3: flash attention per (b,h). Block 128 = 4 waves; 64 query rows;
// key blocks of 64 -> 16 WMMAs per wave per barrier triple.
// ---------------------------------------------------------------------------
__global__ __launch_bounds__(128) void attn_kernel(
    const u16* __restrict__ qg, const u16* __restrict__ kg,
    const u16* __restrict__ vg, u16* __restrict__ ob) {
  __shared__ u16 lq[64 * 72];   // [qrow][hd]
  __shared__ u16 lk[64 * 72];   // [key][hd]
  __shared__ u16 lv[64 * 72];   // [hd][key]   (transposed)
  __shared__ u16 lp[64 * 72];   // [qrow][key]

  const int tid = threadIdx.x, lane = tid & 31, wv = tid >> 5;
  const int half = lane >> 4, col = lane & 15;
  const int bh = blockIdx.y;
  const int q0 = blockIdx.x * 64;
  const u16* qbase = qg + (size_t)bh * SEQ * HD;
  const u16* kbase = kg + (size_t)bh * SEQ * HD;
  const u16* vbase = vg + (size_t)bh * SEQ * HD;

  // stage Q tile 64x64 bf16
#pragma unroll
  for (int r = 0; r < 4; ++r) {
    int p = tid + 128 * r, row = p >> 3, seg = p & 7;
    *(uint4*)(lq + row * 72 + seg * 8) =
        *(const uint4*)(qbase + (size_t)(q0 + row) * HD + seg * 8);
  }
  __syncthreads();
  const v16bf qa0 = load_afrag(lq, wv * 16 + col, 72, 0, lane);
  const v16bf qa1 = load_afrag(lq, wv * 16 + col, 72, 32, lane);

  v8f acc[4];
  float mrow[8], lrow[8];
#pragma unroll
  for (int t = 0; t < 4; ++t) acc[t] = (v8f){0.f,0.f,0.f,0.f,0.f,0.f,0.f,0.f};
#pragma unroll
  for (int e = 0; e < 8; ++e) { mrow[e] = -1e30f; lrow[e] = 0.f; }

  for (int kbk = 0; kbk < SEQ / 64; ++kbk) {
    if (kbk + 1 < SEQ / 64) {  // prefetch next K/V block (one line per thread)
      __builtin_prefetch(kbase + (size_t)(kbk + 1) * 64 * HD + tid * 32, 0, 0);
      __builtin_prefetch(vbase + (size_t)(kbk + 1) * 64 * HD + tid * 32, 0, 0);
    }
    // stage K tile (64 keys x 64 hd) as [key][hd]
#pragma unroll
    for (int r = 0; r < 4; ++r) {
      int p = tid + 128 * r, row = p >> 3, seg = p & 7;
      *(uint4*)(lk + row * 72 + seg * 8) =
          *(const uint4*)(kbase + (size_t)(kbk * 64 + row) * HD + seg * 8);
    }
    // stage V tile transposed -> [hd][key]
#pragma unroll
    for (int r = 0; r < 4; ++r) {
      int p = tid + 128 * r, key = p >> 3, hd0 = (p & 7) * 8;
      uint4 t4 = *(const uint4*)(vbase + (size_t)(kbk * 64 + key) * HD + hd0);
      const u16* te = (const u16*)&t4;
#pragma unroll
      for (int i = 0; i < 8; ++i) lv[(hd0 + i) * 72 + key] = te[i];
    }
    __syncthreads();

    // scores: s = q (16x64) @ k^T (64x64), four 16-key n-tiles
    v8f s[4];
#pragma unroll
    for (int t = 0; t < 4; ++t) {
      s[t] = (v8f){0.f,0.f,0.f,0.f,0.f,0.f,0.f,0.f};
      s[t] = wmma_bf16(qa0, load_bfrag(lk, t * 16, 72, 0, lane), s[t]);
      s[t] = wmma_bf16(qa1, load_bfrag(lk, t * 16, 72, 32, lane), s[t]);
    }

    // online softmax per query row (row r = e + 8*half spans 16 lanes)
#pragma unroll
    for (int e = 0; e < 8; ++e) {
      float v0 = s[0][e] * 0.125f, v1 = s[1][e] * 0.125f;
      float v2 = s[2][e] * 0.125f, v3 = s[3][e] * 0.125f;
      float mx = fmaxf(fmaxf(v0, v1), fmaxf(v2, v3));
#pragma unroll
      for (int mask = 8; mask >= 1; mask >>= 1) mx = fmaxf(mx, __shfl_xor(mx, mask, 32));
      float mnew = fmaxf(mrow[e], mx);
      float alpha = __expf(mrow[e] - mnew);
      float p0 = __expf(v0 - mnew), p1 = __expf(v1 - mnew);
      float p2 = __expf(v2 - mnew), p3 = __expf(v3 - mnew);
      float rs = (p0 + p1) + (p2 + p3);
#pragma unroll
      for (int mask = 8; mask >= 1; mask >>= 1) rs += __shfl_xor(rs, mask, 32);
      lrow[e] = lrow[e] * alpha + rs;
      mrow[e] = mnew;
#pragma unroll
      for (int t = 0; t < 4; ++t) acc[t][e] *= alpha;
      int r = e + 8 * half;
      u16* prow = lp + (wv * 16 + r) * 72;
      prow[col]      = f2bf(p0);
      prow[16 + col] = f2bf(p1);
      prow[32 + col] = f2bf(p2);
      prow[48 + col] = f2bf(p3);
    }
    __syncthreads();

    // out += p (16x64) @ v (64x64)
    v16bf pa0 = load_afrag(lp, wv * 16 + col, 72, 0, lane);
    v16bf pa1 = load_afrag(lp, wv * 16 + col, 72, 32, lane);
#pragma unroll
    for (int t = 0; t < 4; ++t) {
      acc[t] = wmma_bf16(pa0, load_bfrag(lv, t * 16, 72, 0, lane), acc[t]);
      acc[t] = wmma_bf16(pa1, load_bfrag(lv, t * 16, 72, 32, lane), acc[t]);
    }
    __syncthreads();
  }

  // epilogue: normalize and write [B,S,D] bf16 (d = h*64 + hd)
  const int bi = bh >> 4, h = bh & 15;
#pragma unroll
  for (int e = 0; e < 8; ++e) {
    int r = e + 8 * half;
    int srow = q0 + wv * 16 + r;
    float inv = 1.0f / lrow[e];
#pragma unroll
    for (int t = 0; t < 4; ++t) {
      int dcol = h * HD + t * 16 + col;
      ob[((size_t)(bi * SEQ + srow)) * DIM + dcol] = f2bf(acc[t][e] * inv);
    }
  }
}

// ---------------------------------------------------------------------------
// Kernel 4: out = attn @ out_w^T + out_b  (fp32 output)
// Block tile 64x64, K-step 64, two M-tiles per wave.
// ---------------------------------------------------------------------------
__global__ __launch_bounds__(256) void out_gemm_kernel(
    const u16* __restrict__ abf, const float* __restrict__ w,
    const float* __restrict__ bias, float* __restrict__ out) {
  __shared__ u16 la[64 * 72];
  __shared__ u16 lb[64 * 72];
  const int tid = threadIdx.x, lane = tid & 31, wv = tid >> 5;
  const int wr = wv >> 2, wc = wv & 3;
  const int m0 = blockIdx.x * 64, n0 = blockIdx.y * 64;

  v8f acc0 = {0.f,0.f,0.f,0.f,0.f,0.f,0.f,0.f};
  v8f acc1 = {0.f,0.f,0.f,0.f,0.f,0.f,0.f,0.f};

  for (int k0 = 0; k0 < DIM; k0 += 64) {
    if (k0 + 64 < DIM) {
      __builtin_prefetch(abf + (size_t)(m0 + (tid >> 2)) * DIM + k0 + 64 + (tid & 3) * 16, 0, 0);
      __builtin_prefetch(w + (size_t)(n0 + (tid >> 2)) * DIM + k0 + 64 + (tid & 3) * 16, 0, 0);
    }
#pragma unroll
    for (int r = 0; r < 4; ++r) {  // stage A (already bf16): 32 rows x 64
      int p = tid + 256 * r, row = p >> 5, c2 = (p & 31) * 2;
      *(u32*)(la + row * 72 + c2) =
          *(const u32*)(abf + (size_t)(m0 + row) * DIM + k0 + c2);
    }
#pragma unroll
    for (int r = 0; r < 4; ++r) {  // stage B (fp32 -> bf16), [n][k]
      int e = tid * 4 + r * 1024, row = e >> 6, colk = e & 63;
      float4 f = *(const float4*)(w + (size_t)(n0 + row) * DIM + k0 + colk);
      uint2 pk;
      pk.x = (u32)f2bf(f.x) | ((u32)f2bf(f.y) << 16);
      pk.y = (u32)f2bf(f.z) | ((u32)f2bf(f.w) << 16);
      *(uint2*)(lb + row * 72 + colk) = pk;
    }
    __syncthreads();
#pragma unroll
    for (int ks = 0; ks < 64; ks += 32) {
      v16bf b  = load_bfrag(lb, wc * 16, 72, ks, lane);
      v16bf a0 = load_afrag(la, wr * 32 + (lane & 15), 72, ks, lane);
      v16bf a1 = load_afrag(la, wr * 32 + 16 + (lane & 15), 72, ks, lane);
      acc0 = wmma_bf16(a0, b, acc0);
      acc1 = wmma_bf16(a1, b, acc1);
    }
    __syncthreads();
  }

  const int half = lane >> 4;
  const int ncol = n0 + wc * 16 + (lane & 15);
  const float bsv = bias[ncol];
#pragma unroll
  for (int e = 0; e < 8; ++e) {
    int m = m0 + wr * 32 + e + 8 * half;
    out[(size_t)m * DIM + ncol] = acc0[e] + bsv;
    out[(size_t)(m + 16) * DIM + ncol] = acc1[e] + bsv;
  }
}

// ---------------------------------------------------------------------------
extern "C" void kernel_launch(void* const* d_in, const int* in_sizes, int n_in,
                              void* d_out, int out_size, void* d_ws, size_t ws_size,
                              hipStream_t stream) {
  const float* x     = (const float*)d_in[0];
  const float* qkv_w = (const float*)d_in[1];
  const float* qkv_b = (const float*)d_in[2];
  const float* q_g   = (const float*)d_in[3];
  const float* q_bt  = (const float*)d_in[4];
  const float* k_g   = (const float*)d_in[5];
  const float* k_bt  = (const float*)d_in[6];
  const float* out_w = (const float*)d_in[7];
  const float* out_b = (const float*)d_in[8];
  float* out = (float*)d_out;

  const size_t bufElems = (size_t)BATCH * HEADS * SEQ * HD;  // 8,388,608
  u16* qb = (u16*)d_ws;
  u16* kb = qb + bufElems;
  u16* vb = kb + bufElems;
  u16* ab = vb + bufElems;   // attention output, [B,S,D] bf16

  qkv_gemm_kernel<<<dim3(MTOT / 64, NQKV / 64), 256, 0, stream>>>(
      x, qkv_w, qkv_b, qb, kb, vb);
  qk_ln_kernel<<<(2 * ROWS_PER_BUF) / 8, 256, 0, stream>>>(
      qb, kb, q_g, q_bt, k_g, k_bt);
  attn_kernel<<<dim3(SEQ / 64, BATCH * HEADS), 128, 0, stream>>>(
      qb, kb, vb, ab);
  out_gemm_kernel<<<dim3(MTOT / 64, DIM / 64), 256, 0, stream>>>(
      ab, out_w, out_b, out);
}
